// DAGModel_51642686767118
// MI455X (gfx1250) — compile-verified, matrix operand order checked by reading, not run on hardware
//
#include <hip/hip_runtime.h>
#include <hip/hip_bf16.h>
#include <math.h>

// Problem constants from the reference.
#define B_  32
#define N_  128
#define ED_ 16
#define NL_ 3

typedef __attribute__((ext_vector_type(16))) _Float16 v16h;
typedef __attribute__((ext_vector_type(8)))  float    v8f;

// CDNA5 16-bit A/B fragment K index for element e (0..15) and lane-group g (0/1).
// A 16x32 f16: V0..3 lanes0-15 K=0..7, lanes16-31 K=8..15; V4..7 K=16..23 / 24..31.
__device__ __forceinline__ int kof(int e, int g) {
  return ((e & 8) << 1) + (g << 3) + (e & 7);
}

__device__ __forceinline__ float sig_(float x) { return 1.0f / (1.0f + __expf(-x)); }

// ---------------------------------------------------------------------------
// h = nodes * w + b   (node_embedding, 1->1)
// ---------------------------------------------------------------------------
__global__ __launch_bounds__(256) void embed_kernel(
    const float* __restrict__ nodes, const float* __restrict__ w,
    const float* __restrict__ b, float* __restrict__ h) {
  int n = blockIdx.x * 256 + threadIdx.x;
  if (n < B_ * N_) h[n] = nodes[n] * w[0] + b[0];
}

// ---------------------------------------------------------------------------
// Per-node precompute for one layer:
//   src_h = relu(h*ws1+bs1) @ Ws2 + bs2      (1->64->16)
//   tgt_h analogously
//   S1[n] = src_h @ Wa1[0:16,:]  + ba1       (16->128, edge_attn L1, src half)
//   T1[n] = tgt_h @ Wa1[16:32,:]             (16->128, tgt half)
//   SA[n] = src_h @ We1                      (16->64,  edge_aggr L1)
//   TA[n] = tgt_h @ We1
// ---------------------------------------------------------------------------
__global__ __launch_bounds__(256) void node_pre_kernel(
    const float* __restrict__ h,
    const float* __restrict__ ws1, const float* __restrict__ bs1,
    const float* __restrict__ ws2, const float* __restrict__ bs2,
    const float* __restrict__ wt1, const float* __restrict__ bt1,
    const float* __restrict__ wt2, const float* __restrict__ bt2,
    const float* __restrict__ wa1, const float* __restrict__ ba1,
    const float* __restrict__ we1,
    float* __restrict__ S1, float* __restrict__ T1,
    float* __restrict__ SA, float* __restrict__ TA) {
  int n = blockIdx.x * 256 + threadIdx.x;
  if (n >= B_ * N_) return;
  float hv = h[n];

  float sh[16], th[16];
#pragma unroll
  for (int o = 0; o < 16; ++o) { sh[o] = bs2[o]; th[o] = bt2[o]; }
  for (int k = 0; k < 64; ++k) {
    float es = fmaxf(hv * ws1[k] + bs1[k], 0.0f);
    float et = fmaxf(hv * wt1[k] + bt1[k], 0.0f);
#pragma unroll
    for (int o = 0; o < 16; ++o) {
      sh[o] += es * ws2[k * 16 + o];
      th[o] += et * wt2[k * 16 + o];
    }
  }

  for (int j = 0; j < 128; ++j) {
    float a = ba1[j];       // fold edge_attn L1 bias into S1
    float t = 0.0f;
#pragma unroll
    for (int k = 0; k < 16; ++k) {
      a += sh[k] * wa1[k * 128 + j];
      t += th[k] * wa1[(16 + k) * 128 + j];
    }
    S1[n * 128 + j] = a;
    T1[n * 128 + j] = t;
  }
  for (int j = 0; j < 64; ++j) {
    float a = 0.0f, t = 0.0f;
#pragma unroll
    for (int k = 0; k < 16; ++k) {
      a += sh[k] * we1[k * 64 + j];
      t += th[k] * we1[k * 64 + j];
    }
    SA[n * 64 + j] = a;      // edge_aggr L1 bias kept separate (scaled by w_ij)
    TA[n * 64 + j] = t;
  }
}

__global__ __launch_bounds__(256) void zero_kernel(float* __restrict__ p, int n) {
  int i = blockIdx.x * 256 + threadIdx.x;
  if (i < n) p[i] = 0.0f;
}

// ---------------------------------------------------------------------------
// Edge kernel: each wave owns (b, i, j0..j0+15) = a 16-edge M-tile.
//  a1  = relu(S1[i] + T1[j])                       (128)   [bias folded]
//  a2  = relu(a1 @ Wa2 + b2)                       WMMA 4x4 (K=128,N=64)
//  attn= a2 . wa3 + b3 ; w = sigmoid(attn)*adj
//  v   = relu(w*(SA[i]+TA[j]) + be1)               (64)
//  v2  = relu(v @ We2 + be2)                       WMMA 2x2 (K=64,N=32)
//  e   = (v2 . we3 + be3)*adj ; node_update[i] += sum_j e
// Block = 8 waves = 8 consecutive i rows sharing LDS-staged weights/tiles.
// ---------------------------------------------------------------------------
__global__ __launch_bounds__(256) void edge_kernel(
    const float* __restrict__ S1, const float* __restrict__ T1,
    const float* __restrict__ SA, const float* __restrict__ TA,
    const float* __restrict__ Wa2, const float* __restrict__ b2,
    const float* __restrict__ wa3, const float* __restrict__ b3,
    const float* __restrict__ be1,
    const float* __restrict__ We2, const float* __restrict__ be2,
    const float* __restrict__ we3, const float* __restrict__ be3,
    const float* __restrict__ adj_logits,
    float* __restrict__ node_update) {
  // Weight B-fragments pre-swizzled to the exact per-lane VGPR layout.
  __shared__ v16h  sWa2f[4 * 4 * 32];   // [ks][nt][lane] 16KB
  __shared__ v16h  sWe2f[2 * 2 * 32];   // [ks][nt][lane] 4KB
  __shared__ float sT1[16 * 128];       // 8KB   [j_local][k]
  __shared__ float sTA[16 * 64];        // 4KB
  __shared__ float sS1[8 * 128];        // 4KB   [wave][k]
  __shared__ float sSA[8 * 64];         // 2KB
  __shared__ float sB2[64], sWa3[64], sBe1[64], sBe2[32], sWe3[32];
  __shared__ __align__(32) _Float16 sA2[8 * 1024];  // 16KB a2 staging / reduce scratch

  const int tid  = threadIdx.x;
  const int wv   = tid >> 5;
  const int lane = tid & 31;
  const int g    = lane >> 4;
  const int ln   = lane & 15;
  const int jb = blockIdx.x, ib = blockIdx.y, b = blockIdx.z;
  const int i  = ib * 8 + wv;
  const int j0 = jb * 16;

  {
    _Float16* p = (_Float16*)sWa2f;
    for (int idx = tid; idx < 8192; idx += 256) {
      int e = idx & 15, l2 = (idx >> 4) & 31, nt = (idx >> 9) & 3, ks = idx >> 11;
      int k = ks * 32 + kof(e, l2 >> 4);
      int n = nt * 16 + (l2 & 15);
      p[idx] = (_Float16)Wa2[k * 64 + n];
    }
    _Float16* q = (_Float16*)sWe2f;
    for (int idx = tid; idx < 2048; idx += 256) {
      int e = idx & 15, l2 = (idx >> 4) & 31, nt = (idx >> 9) & 1, ks = idx >> 10;
      int k = ks * 32 + kof(e, l2 >> 4);
      int n = nt * 16 + (l2 & 15);
      q[idx] = (_Float16)We2[k * 32 + n];
    }
    for (int t = tid; t < 16 * 128; t += 256)
      sT1[t] = T1[(b * N_ + j0 + (t >> 7)) * 128 + (t & 127)];
    for (int t = tid; t < 16 * 64; t += 256)
      sTA[t] = TA[(b * N_ + j0 + (t >> 6)) * 64 + (t & 63)];
    for (int t = tid; t < 8 * 128; t += 256)
      sS1[t] = S1[(b * N_ + ib * 8 + (t >> 7)) * 128 + (t & 127)];
    for (int t = tid; t < 8 * 64; t += 256)
      sSA[t] = SA[(b * N_ + ib * 8 + (t >> 6)) * 64 + (t & 63)];
    if (tid < 64) { sB2[tid] = b2[tid]; sWa3[tid] = wa3[tid]; sBe1[tid] = be1[tid]; }
    else if (tid < 96) { sBe2[tid - 64] = be2[tid - 64]; sWe3[tid - 64] = we3[tid - 64]; }
  }
  __syncthreads();

  // adjacency bit for this lane's edge (i, j0+ln): sigmoid(x)>0.5 <=> x>0, upper tri.
  const int jj = j0 + ln;
  const float adjv = (adj_logits[i * N_ + jj] > 0.0f && jj > i) ? 1.0f : 0.0f;

  // ---- GEMM1: (16 x 128) @ (128 x 64) via 16 WMMAs -------------------------
  v8f acc[4] = {v8f{}, v8f{}, v8f{}, v8f{}};
#pragma unroll
  for (int ks = 0; ks < 4; ++ks) {
    v16h A;
#pragma unroll
    for (int e = 0; e < 16; ++e) {
      int k = ks * 32 + kof(e, g);
      A[e] = (_Float16)fmaxf(sS1[wv * 128 + k] + sT1[ln * 128 + k], 0.0f);
    }
#pragma unroll
    for (int nt = 0; nt < 4; ++nt) {
      acc[nt] = __builtin_amdgcn_wmma_f32_16x16x32_f16(
          false, A, false, sWa2f[(ks * 4 + nt) * 32 + lane],
          (short)0, acc[nt], false, false);
    }
  }

  // a2 = relu(acc + b2), stage as f16 [edge][n] per wave.
  _Float16* myA2 = &sA2[wv * 1024];
#pragma unroll
  for (int nt = 0; nt < 4; ++nt)
#pragma unroll
    for (int r = 0; r < 8; ++r) {
      int n = nt * 16 + ln;
      int e = r + 8 * g;
      myA2[e * 64 + n] = (_Float16)fmaxf(acc[nt][r] + sB2[n], 0.0f);
    }
  __syncthreads();

  // attn head: each lane does half the 64-dot for edge ln, combine halves.
  float part = 0.0f;
#pragma unroll
  for (int k = 0; k < 32; ++k) {
    int kk = g * 32 + k;
    part += (float)myA2[ln * 64 + kk] * sWa3[kk];
  }
  part += __shfl_xor(part, 16, 32);
  const float w = adjv * sig_(part + b3[0]);
  __syncthreads();  // everyone done reading sA2 before it is reused

  // ---- GEMM2: (16 x 64) @ (64 x 32) via 4 WMMAs ----------------------------
  v8f c2[2] = {v8f{}, v8f{}};
#pragma unroll
  for (int ks = 0; ks < 2; ++ks) {
    v16h A;
#pragma unroll
    for (int e = 0; e < 16; ++e) {
      int k = ks * 32 + kof(e, g);
      float x = w * (sSA[wv * 64 + k] + sTA[ln * 64 + k]) + sBe1[k];
      A[e] = (_Float16)fmaxf(x, 0.0f);
    }
#pragma unroll
    for (int nt = 0; nt < 2; ++nt) {
      c2[nt] = __builtin_amdgcn_wmma_f32_16x16x32_f16(
          false, A, false, sWe2f[(ks * 2 + nt) * 32 + lane],
          (short)0, c2[nt], false, false);
    }
  }

  // aggr head: relu(c2+be2).we3, reduce 16-lane-spread dot per edge via LDS.
  float* red = (float*)&sA2[wv * 1024];  // 256 floats per wave
#pragma unroll
  for (int r = 0; r < 8; ++r) {
    int e = r + 8 * g;
    float p = 0.0f;
#pragma unroll
    for (int nt = 0; nt < 2; ++nt) {
      int n = nt * 16 + ln;
      p += fmaxf(c2[nt][r] + sBe2[n], 0.0f) * sWe3[n];
    }
    red[e * 16 + ln] = p;
  }
  __syncthreads();

  float s = 0.0f;
#pragma unroll
  for (int c = 0; c < 8; ++c) s += red[ln * 16 + g * 8 + c];
  s += __shfl_xor(s, 16, 32);                 // full 16-wide dot for edge ln
  float ev = (s + be3[0]) * adjv;             // e_aggr masked by adj
  ev += __shfl_xor(ev, 1, 32);                // sum over the 16 j's of this tile
  ev += __shfl_xor(ev, 2, 32);
  ev += __shfl_xor(ev, 4, 32);
  ev += __shfl_xor(ev, 8, 32);
  if (lane == 0) atomicAdd(&node_update[b * N_ + i], ev);
}

// ---------------------------------------------------------------------------
// Node update: MLP([h, nu]) (2->64->32->1) + sigmoid(skip)*h, in place.
// ---------------------------------------------------------------------------
__global__ __launch_bounds__(256) void node_upd_kernel(
    float* __restrict__ h, const float* __restrict__ nu,
    const float* __restrict__ wn1, const float* __restrict__ bn1,
    const float* __restrict__ wn2, const float* __restrict__ bn2,
    const float* __restrict__ wn3, const float* __restrict__ bn3,
    const float* __restrict__ skip) {
  int n = blockIdx.x * 256 + threadIdx.x;
  if (n >= B_ * N_) return;
  float h0 = h[n], u = nu[n];
  float t1[64];
  for (int k = 0; k < 64; ++k)
    t1[k] = fmaxf(h0 * wn1[k] + u * wn1[64 + k] + bn1[k], 0.0f);
  float tr = bn3[0];
  for (int o = 0; o < 32; ++o) {
    float a = bn2[o];
    for (int k = 0; k < 64; ++k) a += t1[k] * wn2[k * 32 + o];
    tr += fmaxf(a, 0.0f) * wn3[o];
  }
  h[n] = tr + sig_(skip[0]) * h0;
}

// ---------------------------------------------------------------------------
// Output head: relu(h*w0+b0)*w1+b1 -> d_out[0:4096]
// Adjacency:   d_out[4096 + i*N + j]
// ---------------------------------------------------------------------------
__global__ __launch_bounds__(256) void output_kernel(
    const float* __restrict__ h,
    const float* __restrict__ w0, const float* __restrict__ b0,
    const float* __restrict__ w1, const float* __restrict__ b1,
    float* __restrict__ out) {
  int n = blockIdx.x * 256 + threadIdx.x;
  if (n < B_ * N_) out[n] = fmaxf(h[n] * w0[0] + b0[0], 0.0f) * w1[0] + b1[0];
}

__global__ __launch_bounds__(256) void adj_kernel(
    const float* __restrict__ logits, float* __restrict__ out) {
  int idx = blockIdx.x * 256 + threadIdx.x;
  if (idx >= N_ * N_) return;
  int i = idx >> 7, j = idx & 127;
  out[idx] = (logits[idx] > 0.0f && j > i) ? 1.0f : 0.0f;
}

// ---------------------------------------------------------------------------
extern "C" void kernel_launch(void* const* d_in, const int* in_sizes, int n_in,
                              void* d_out, int out_size, void* d_ws, size_t ws_size,
                              hipStream_t stream) {
  (void)in_sizes; (void)n_in; (void)out_size; (void)ws_size;
  const float* nodes = (const float*)d_in[0];
  const float* adjl  = (const float*)d_in[1];
  const float* we_w  = (const float*)d_in[2];
  const float* we_b  = (const float*)d_in[3];
  const float* ws1 = (const float*)d_in[4],  *bs1 = (const float*)d_in[5];
  const float* ws2 = (const float*)d_in[6],  *bs2 = (const float*)d_in[7];
  const float* wt1 = (const float*)d_in[8],  *bt1 = (const float*)d_in[9];
  const float* wt2 = (const float*)d_in[10], *bt2 = (const float*)d_in[11];
  const float* wa1 = (const float*)d_in[12], *ba1 = (const float*)d_in[13];
  const float* wa2 = (const float*)d_in[14], *ba2 = (const float*)d_in[15];
  const float* wa3 = (const float*)d_in[16], *ba3 = (const float*)d_in[17];
  const float* wg1 = (const float*)d_in[18], *bg1 = (const float*)d_in[19];
  const float* wg2 = (const float*)d_in[20], *bg2 = (const float*)d_in[21];
  const float* wg3 = (const float*)d_in[22], *bg3 = (const float*)d_in[23];
  const float* wn1 = (const float*)d_in[24], *bn1 = (const float*)d_in[25];
  const float* wn2 = (const float*)d_in[26], *bn2 = (const float*)d_in[27];
  const float* wn3 = (const float*)d_in[28], *bn3 = (const float*)d_in[29];
  const float* skip = (const float*)d_in[30];
  const float* ow0 = (const float*)d_in[31], *ob0 = (const float*)d_in[32];
  const float* ow1 = (const float*)d_in[33], *ob1 = (const float*)d_in[34];
  float* out = (float*)d_out;

  float* ws = (float*)d_ws;
  float* S1 = ws;                    // B*N*128
  float* T1 = S1 + B_ * N_ * 128;    // B*N*128
  float* SA = T1 + B_ * N_ * 128;    // B*N*64
  float* TA = SA + B_ * N_ * 64;     // B*N*64
  float* h  = TA + B_ * N_ * 64;     // B*N
  float* nu = h + B_ * N_;           // B*N

  const int nodeBlocks = (B_ * N_) / 256;  // 16
  embed_kernel<<<nodeBlocks, 256, 0, stream>>>(nodes, we_w, we_b, h);

  dim3 egrid(N_ / 16, N_ / 8, B_);  // (8, 16, 32)
  for (int L = 0; L < NL_; ++L) {
    node_pre_kernel<<<nodeBlocks, 256, 0, stream>>>(
        h, ws1, bs1, ws2, bs2, wt1, bt1, wt2, bt2, wa1, ba1, wg1,
        S1, T1, SA, TA);
    zero_kernel<<<nodeBlocks, 256, 0, stream>>>(nu, B_ * N_);
    edge_kernel<<<egrid, 256, 0, stream>>>(
        S1, T1, SA, TA, wa2, ba2, wa3, ba3, bg1, wg2, bg2, wg3, bg3,
        adjl, nu);
    node_upd_kernel<<<nodeBlocks, 256, 0, stream>>>(
        h, nu, wn1, bn1, wn2, bn2, wn3, bn3, skip);
  }

  output_kernel<<<nodeBlocks, 256, 0, stream>>>(h, ow0, ob0, ow1, ob1, out);
  adj_kernel<<<(N_ * N_) / 256, 256, 0, stream>>>(adjl, out + B_ * N_);
}
